// resGCNmodel_7310034338106
// MI455X (gfx1250) — compile-verified, matrix operand order checked by reading, not run on hardware
//
#include <hip/hip_runtime.h>
#include <hip/hip_bf16.h>

#define N_NODES 50000
#define N_EDGES 600000
#define F_INPUT 512
#define HDIM    128
#define CDIM    16
#define NLAYERS 8
#define ALPHA_C 0.9f
#define GAMMA_C 0.1f
#define LAMDA_C 0.5f

typedef __attribute__((ext_vector_type(16))) __bf16 v16bf;
typedef __attribute__((ext_vector_type(8)))  float  v8f;

union AFrag { v16bf v; unsigned int u[8]; };

__device__ __forceinline__ unsigned int pack2_bf16(float lo, float hi) {
    unsigned int a = __float_as_uint(lo);
    unsigned int b = __float_as_uint(hi);
    a = (a + 0x7FFFu + ((a >> 16) & 1u)) >> 16;   // RNE f32 -> bf16
    b = (b + 0x7FFFu + ((b >> 16) & 1u)) >> 16;
    return (b << 16) | (a & 0xFFFFu);
}

__device__ __forceinline__ unsigned short cvt_bf16(float f) {
    unsigned int a = __float_as_uint(f);
    a = (a + 0x7FFFu + ((a >> 16) & 1u)) >> 16;
    return (unsigned short)a;
}

// ---------------------------------------------------------------- utilities
__global__ void k_zero(float* __restrict__ p, int n) {
    int i = blockIdx.x * blockDim.x + threadIdx.x;
    if (i < n) p[i] = 0.f;
}

// WT1[n][k] = (n<128 ? W0[k][n] : Whp[k][n-128]), bf16, n in [0,256), k in [0,512)
__global__ void k_prep_w1(const float* __restrict__ W0, const float* __restrict__ Whp,
                          unsigned short* __restrict__ WT1) {
    int idx = blockIdx.x * blockDim.x + threadIdx.x;
    if (idx >= 256 * F_INPUT) return;
    int n = idx / F_INPUT, k = idx % F_INPUT;
    float v = (n < HDIM) ? W0[(size_t)k * HDIM + n] : Whp[(size_t)k * HDIM + (n - HDIM)];
    WT1[idx] = cvt_bf16(v);
}

// WlT[i][n][k] = Wl[i][k][n] bf16
__global__ void k_prep_wl(const float* __restrict__ Wl, unsigned short* __restrict__ WlT) {
    int idx = blockIdx.x * blockDim.x + threadIdx.x;
    if (idx >= NLAYERS * HDIM * HDIM) return;
    int i = idx / (HDIM * HDIM);
    int r = idx % (HDIM * HDIM);
    int n = r / HDIM, k = r % HDIM;
    WlT[idx] = cvt_bf16(Wl[(size_t)i * HDIM * HDIM + (size_t)k * HDIM + n]);
}

// WoutT[n][k] = Wout[k][n] bf16, n<16, k<128
__global__ void k_prep_wout(const float* __restrict__ Wout, unsigned short* __restrict__ WoutT) {
    int idx = blockIdx.x * blockDim.x + threadIdx.x;
    if (idx >= CDIM * HDIM) return;
    int n = idx / HDIM, k = idx % HDIM;
    WoutT[idx] = cvt_bf16(Wout[(size_t)k * CDIM + n]);
}

// ---------------------------------------------------------------- degrees
__global__ void k_degrees(const long long* __restrict__ ei,
                          float* __restrict__ deg, float* __restrict__ outdeg) {
    int e = blockIdx.x * blockDim.x + threadIdx.x;
    if (e >= N_EDGES) return;
    int s = (int)ei[e];
    int d = (int)ei[N_EDGES + e];
    atomicAdd(&outdeg[s], 1.f);
    atomicAdd(&deg[d], 1.f);
}

__global__ void k_dinv(const float* __restrict__ deg, float* __restrict__ dinv) {
    int i = blockIdx.x * blockDim.x + threadIdx.x;
    if (i < N_NODES) dinv[i] = rsqrtf(deg[i] + 1.0f);
}

// ------------------------------------------------- input GEMM: h0 & y (WMMA)
// One wave per (16-row tile, half): computes 8 16x16 tiles over K=512.
__global__ void __launch_bounds__(256)
k_gemm_in(const float* __restrict__ x, const unsigned short* __restrict__ WT1,
          const float* __restrict__ b0, float* __restrict__ h0, float* __restrict__ y) {
    const int lane = threadIdx.x & 31;
    const int gw   = (blockIdx.x * 256 + threadIdx.x) >> 5;
    const int row_tile = gw >> 1;
    const int half     = gw & 1;                 // 0 -> cols 0..127 (h0), 1 -> 128..255 (y)
    if (row_tile >= N_NODES / 16) return;
    const int m0 = row_tile * 16;
    const int lh = lane >> 4;                    // lane half (K sub-chunk select)
    const int ln = lane & 15;
    const int arow = m0 + ln;

    v8f zero = {0.f,0.f,0.f,0.f,0.f,0.f,0.f,0.f};
    v8f acc[8];
#pragma unroll
    for (int j = 0; j < 8; ++j) acc[j] = zero;

    const float* xrow = x + (size_t)arow * F_INPUT;
    for (int kk = 0; kk < F_INPUT; kk += 32) {
        const int k0 = kk + lh * 8;
        AFrag a;
        float4 q0 = *(const float4*)(xrow + k0);
        float4 q1 = *(const float4*)(xrow + k0 + 4);
        float4 q2 = *(const float4*)(xrow + k0 + 16);
        float4 q3 = *(const float4*)(xrow + k0 + 20);
        a.u[0] = pack2_bf16(q0.x, q0.y); a.u[1] = pack2_bf16(q0.z, q0.w);
        a.u[2] = pack2_bf16(q1.x, q1.y); a.u[3] = pack2_bf16(q1.z, q1.w);
        a.u[4] = pack2_bf16(q2.x, q2.y); a.u[5] = pack2_bf16(q2.z, q2.w);
        a.u[6] = pack2_bf16(q3.x, q3.y); a.u[7] = pack2_bf16(q3.z, q3.w);
#pragma unroll
        for (int j = 0; j < 8; ++j) {
            const int ncol = half * 128 + j * 16 + ln;
            const uint4* bp = (const uint4*)(WT1 + (size_t)ncol * F_INPUT + k0);
            uint4 b0v = bp[0];                   // K = k0..k0+7
            uint4 b1v = bp[2];                   // K = k0+16..k0+23
            AFrag b;
            b.u[0]=b0v.x; b.u[1]=b0v.y; b.u[2]=b0v.z; b.u[3]=b0v.w;
            b.u[4]=b1v.x; b.u[5]=b1v.y; b.u[6]=b1v.z; b.u[7]=b1v.w;
            acc[j] = __builtin_amdgcn_wmma_f32_16x16x32_bf16(
                         false, a.v, false, b.v, (short)0, acc[j], false, false);
        }
    }

#pragma unroll
    for (int j = 0; j < 8; ++j) {
        const int c = j * 16 + ln;               // column within 128-wide half
        if (half == 0) {
            float bias = b0[c];
#pragma unroll
            for (int r = 0; r < 8; ++r) {
                int row = m0 + lh * 8 + r;
                float v = acc[j][r] + bias;
                h0[(size_t)row * HDIM + c] = v > 0.f ? v : 0.f;
            }
        } else {
#pragma unroll
            for (int r = 0; r < 8; ++r) {
                int row = m0 + lh * 8 + r;
                y[(size_t)row * HDIM + c] = acc[j][r];
            }
        }
    }
}

// --------------------------------------------- edge scatter kernels (1 edge/wave)
__global__ void k_edge_agg_hp(const long long* __restrict__ ei,
                              const float* __restrict__ y, float* __restrict__ agg) {
    int e = (blockIdx.x * 256 + threadIdx.x) >> 5;
    if (e >= N_EDGES) return;
    int lane = threadIdx.x & 31;
    int s = (int)ei[e];
    int d = (int)ei[N_EDGES + e];
    float4 v = *(const float4*)(y + (size_t)s * HDIM + lane * 4);
    float* dst = agg + (size_t)d * HDIM + lane * 4;
    atomicAdd(dst + 0, v.x); atomicAdd(dst + 1, v.y);
    atomicAdd(dst + 2, v.z); atomicAdd(dst + 3, v.w);
}

__global__ void k_edge_agg_prop(const long long* __restrict__ ei,
                                const float* __restrict__ h,
                                const float* __restrict__ dinv, float* __restrict__ agg) {
    int e = (blockIdx.x * 256 + threadIdx.x) >> 5;
    if (e >= N_EDGES) return;
    int lane = threadIdx.x & 31;
    int s = (int)ei[e];
    int d = (int)ei[N_EDGES + e];
    float w = dinv[s] * dinv[d];
    float4 v = *(const float4*)(h + (size_t)s * HDIM + lane * 4);
    float* dst = agg + (size_t)d * HDIM + lane * 4;
    atomicAdd(dst + 0, w * v.x); atomicAdd(dst + 1, w * v.y);
    atomicAdd(dst + 2, w * v.z); atomicAdd(dst + 3, w * v.w);
}

// lx = relu(outdeg*y - agg + bhp)
__global__ void k_lx(const float* __restrict__ y, const float* __restrict__ agg,
                     const float* __restrict__ outdeg, const float* __restrict__ bhp,
                     float* __restrict__ lx) {
    int i = blockIdx.x * 256 + threadIdx.x;
    if (i >= N_NODES * HDIM) return;
    int node = i >> 7, f = i & 127;
    float v = outdeg[node] * y[i] - agg[i] + bhp[f];
    lx[i] = v > 0.f ? v : 0.f;
}

// support = ALPHA*(agg + dinv^2*h) + (1-ALPHA)*h0 + GAMMA*lx ; f32 + bf16 copies
__global__ void k_support(const float* __restrict__ h, const float* __restrict__ h0,
                          const float* __restrict__ lx, const float* __restrict__ agg,
                          const float* __restrict__ dinv,
                          float* __restrict__ supp, unsigned int* __restrict__ suppbf) {
    int t = blockIdx.x * 256 + threadIdx.x;       // 2 features per thread
    if (t >= N_NODES * HDIM / 2) return;
    int node = t >> 6;
    int f2 = (t & 63) << 1;
    size_t idx = (size_t)node * HDIM + f2;
    float dn = dinv[node]; float sn = dn * dn;
    float2 hh = *(const float2*)(h   + idx);
    float2 h0v= *(const float2*)(h0  + idx);
    float2 ll = *(const float2*)(lx  + idx);
    float2 aa = *(const float2*)(agg + idx);
    float s0 = ALPHA_C * (aa.x + sn * hh.x) + (1.f - ALPHA_C) * h0v.x + GAMMA_C * ll.x;
    float s1 = ALPHA_C * (aa.y + sn * hh.y) + (1.f - ALPHA_C) * h0v.y + GAMMA_C * ll.y;
    float2 so; so.x = s0; so.y = s1;
    *(float2*)(supp + idx) = so;
    suppbf[idx >> 1] = pack2_bf16(s0, s1);
}

// ------------------------------------------- layer GEMM: relu((1-b)S + b*(S@Wl))
__global__ void __launch_bounds__(256)
k_gemm_layer(const unsigned int* __restrict__ suppbf, const float* __restrict__ supp,
             const unsigned short* __restrict__ WlTi, float beta, float* __restrict__ hout) {
    const int lane = threadIdx.x & 31;
    const int gw = (blockIdx.x * 256 + threadIdx.x) >> 5;
    if (gw >= N_NODES / 16) return;
    const int m0 = gw * 16;
    const int lh = lane >> 4, ln = lane & 15;
    const int arow = m0 + ln;

    v8f zero = {0.f,0.f,0.f,0.f,0.f,0.f,0.f,0.f};
    v8f acc[8];
#pragma unroll
    for (int j = 0; j < 8; ++j) acc[j] = zero;

    const unsigned short* sb = (const unsigned short*)suppbf;
    for (int kk = 0; kk < HDIM; kk += 32) {
        const int k0 = kk + lh * 8;
        const uint4* ap = (const uint4*)(sb + (size_t)arow * HDIM + k0);
        uint4 a0 = ap[0], a1 = ap[2];
        AFrag a;
        a.u[0]=a0.x; a.u[1]=a0.y; a.u[2]=a0.z; a.u[3]=a0.w;
        a.u[4]=a1.x; a.u[5]=a1.y; a.u[6]=a1.z; a.u[7]=a1.w;
#pragma unroll
        for (int j = 0; j < 8; ++j) {
            const int ncol = j * 16 + ln;
            const uint4* bp = (const uint4*)(WlTi + (size_t)ncol * HDIM + k0);
            uint4 b0v = bp[0], b1v = bp[2];
            AFrag b;
            b.u[0]=b0v.x; b.u[1]=b0v.y; b.u[2]=b0v.z; b.u[3]=b0v.w;
            b.u[4]=b1v.x; b.u[5]=b1v.y; b.u[6]=b1v.z; b.u[7]=b1v.w;
            acc[j] = __builtin_amdgcn_wmma_f32_16x16x32_bf16(
                         false, a.v, false, b.v, (short)0, acc[j], false, false);
        }
    }

    const float omb = 1.f - beta;
#pragma unroll
    for (int j = 0; j < 8; ++j) {
        const int c = j * 16 + ln;
#pragma unroll
        for (int r = 0; r < 8; ++r) {
            int row = m0 + lh * 8 + r;
            size_t idx = (size_t)row * HDIM + c;
            float v = omb * supp[idx] + beta * acc[j][r];
            hout[idx] = v > 0.f ? v : 0.f;
        }
    }
}

// --------------------------- output GEMM + log_softmax (C=16 == one WMMA tile)
__global__ void __launch_bounds__(256)
k_gemm_out(const float* __restrict__ h, const unsigned short* __restrict__ WoutT,
           const float* __restrict__ bout, float* __restrict__ out) {
    const int lane = threadIdx.x & 31;
    const int gw = (blockIdx.x * 256 + threadIdx.x) >> 5;
    if (gw >= N_NODES / 16) return;
    const int m0 = gw * 16;
    const int lh = lane >> 4, ln = lane & 15;
    const int arow = m0 + ln;

    v8f acc = {0.f,0.f,0.f,0.f,0.f,0.f,0.f,0.f};
    const float* hrow = h + (size_t)arow * HDIM;
    for (int kk = 0; kk < HDIM; kk += 32) {
        const int k0 = kk + lh * 8;
        AFrag a;
        float4 q0 = *(const float4*)(hrow + k0);
        float4 q1 = *(const float4*)(hrow + k0 + 4);
        float4 q2 = *(const float4*)(hrow + k0 + 16);
        float4 q3 = *(const float4*)(hrow + k0 + 20);
        a.u[0] = pack2_bf16(q0.x, q0.y); a.u[1] = pack2_bf16(q0.z, q0.w);
        a.u[2] = pack2_bf16(q1.x, q1.y); a.u[3] = pack2_bf16(q1.z, q1.w);
        a.u[4] = pack2_bf16(q2.x, q2.y); a.u[5] = pack2_bf16(q2.z, q2.w);
        a.u[6] = pack2_bf16(q3.x, q3.y); a.u[7] = pack2_bf16(q3.z, q3.w);
        const uint4* bp = (const uint4*)(WoutT + (size_t)ln * HDIM + k0);
        uint4 b0v = bp[0], b1v = bp[2];
        AFrag b;
        b.u[0]=b0v.x; b.u[1]=b0v.y; b.u[2]=b0v.z; b.u[3]=b0v.w;
        b.u[4]=b1v.x; b.u[5]=b1v.y; b.u[6]=b1v.z; b.u[7]=b1v.w;
        acc = __builtin_amdgcn_wmma_f32_16x16x32_bf16(
                  false, a.v, false, b.v, (short)0, acc, false, false);
    }

    const float bias = bout[ln];
#pragma unroll
    for (int r = 0; r < 8; ++r) {
        float v = acc[r] + bias;
        float m = v;                              // row-max across 16 lanes
#pragma unroll
        for (int off = 1; off < 16; off <<= 1)
            m = fmaxf(m, __shfl_xor(m, off, 32));
        float s = __expf(v - m);                  // row sum(exp)
#pragma unroll
        for (int off = 1; off < 16; off <<= 1)
            s += __shfl_xor(s, off, 32);
        float res = (v - m) - __logf(s);
        int row = m0 + lh * 8 + r;
        out[(size_t)row * CDIM + ln] = res;
    }
}

// ---------------------------------------------------------------- launcher
extern "C" void kernel_launch(void* const* d_in, const int* in_sizes, int n_in,
                              void* d_out, int out_size, void* d_ws, size_t ws_size,
                              hipStream_t stream) {
    (void)in_sizes; (void)n_in; (void)out_size; (void)ws_size;
    const float*     x    = (const float*)d_in[0];
    const long long* ei   = (const long long*)d_in[1];
    const float*     W0   = (const float*)d_in[2];
    const float*     b0   = (const float*)d_in[3];
    const float*     Whp  = (const float*)d_in[4];
    const float*     bhp  = (const float*)d_in[5];
    const float*     Wl   = (const float*)d_in[6];
    const float*     Wout = (const float*)d_in[7];
    const float*     bout = (const float*)d_in[8];

    char* ws = (char*)d_ws;
    size_t off = 0;
    auto alloc = [&](size_t bytes) -> char* {
        char* p = ws + off;
        off += (bytes + 255) & ~(size_t)255;
        return p;
    };
    const size_t NH = (size_t)N_NODES * HDIM;
    unsigned short* WT1    = (unsigned short*)alloc((size_t)256 * F_INPUT * 2);
    unsigned short* WlT    = (unsigned short*)alloc((size_t)NLAYERS * HDIM * HDIM * 2);
    unsigned short* WoutT  = (unsigned short*)alloc((size_t)CDIM * HDIM * 2);
    float* deg    = (float*)alloc((size_t)N_NODES * 4);
    float* outdeg = (float*)alloc((size_t)N_NODES * 4);
    float* dinv   = (float*)alloc((size_t)N_NODES * 4);
    float* h0     = (float*)alloc(NH * 4);
    float* y      = (float*)alloc(NH * 4);
    float* lx     = (float*)alloc(NH * 4);
    float* agg    = (float*)alloc(NH * 4);
    float* supp   = (float*)alloc(NH * 4);
    unsigned int* suppbf = (unsigned int*)alloc(NH * 2);
    float* hA     = (float*)alloc(NH * 4);
    float* hB     = (float*)alloc(NH * 4);

    const int ZB = (int)((NH + 255) / 256);          // zero blocks for N*H
    const int EB = (N_EDGES * 32 + 255) / 256;       // edge-wave blocks
    const int GB1 = ((N_NODES / 16) * 2 + 7) / 8;    // input-GEMM blocks (8 waves/block)
    const int GB  = ((N_NODES / 16) + 7) / 8;        // layer/out GEMM blocks

    // weight prep
    k_prep_w1  <<<(256 * F_INPUT + 255) / 256, 256, 0, stream>>>(W0, Whp, WT1);
    k_prep_wl  <<<(NLAYERS * HDIM * HDIM + 255) / 256, 256, 0, stream>>>(Wl, WlT);
    k_prep_wout<<<(CDIM * HDIM + 255) / 256, 256, 0, stream>>>(Wout, WoutT);

    // degrees + dinv
    k_zero<<<(N_NODES + 255) / 256, 256, 0, stream>>>(deg, N_NODES);
    k_zero<<<(N_NODES + 255) / 256, 256, 0, stream>>>(outdeg, N_NODES);
    k_degrees<<<(N_EDGES + 255) / 256, 256, 0, stream>>>(ei, deg, outdeg);
    k_dinv<<<(N_NODES + 255) / 256, 256, 0, stream>>>(deg, dinv);

    // h0 = relu(x@W0+b0), y = x@Whp (fused WMMA GEMM)
    k_gemm_in<<<GB1, 256, 0, stream>>>(x, WT1, b0, h0, y);

    // high-pass branch: lx = relu(outdeg*y - segsum(y[src]->dst) + bhp)
    k_zero<<<ZB, 256, 0, stream>>>(agg, (int)NH);
    k_edge_agg_hp<<<EB, 256, 0, stream>>>(ei, y, agg);
    k_lx<<<ZB, 256, 0, stream>>>(y, agg, outdeg, bhp, lx);

    // GCNII layers
    const float* hcur = h0;
    float* bufs[2] = { hA, hB };
    for (int i = 0; i < NLAYERS; ++i) {
        float beta = LAMDA_C / (float)(i + 2);
        k_zero<<<ZB, 256, 0, stream>>>(agg, (int)NH);
        k_edge_agg_prop<<<EB, 256, 0, stream>>>(ei, hcur, dinv, agg);
        k_support<<<(int)((NH / 2 + 255) / 256), 256, 0, stream>>>(
            hcur, h0, lx, agg, dinv, supp, suppbf);
        k_gemm_layer<<<GB, 256, 0, stream>>>(
            suppbf, supp, WlT + (size_t)i * HDIM * HDIM, beta, bufs[i & 1]);
        hcur = bufs[i & 1];
    }

    // logits + log_softmax
    k_gemm_out<<<GB, 256, 0, stream>>>(hcur, WoutT, bout, (float*)d_out);
}
